// GATv2Model_82205674045443
// MI455X (gfx1250) — matrix operand order repaired and checked by reading, not compiled, not run on hardware
//
#include <hip/hip_runtime.h>
#include <hip/hip_bf16.h>

// ---------------- problem constants ----------------
#define NM 50000      // nodes
#define EE 800000     // raw edges
#define ETOT 850000   // edges + self loops
#define F1 128        // input features
#define D1 256        // layer-1 output width (8 heads x 32)
#define D2 40         // layer-2 output width (classes)
#define D2P 48        // padded to 3x16 WMMA tiles
#define H1 8          // heads layer 1
#define NEG_SLOPE 0.2f

typedef __attribute__((ext_vector_type(16))) __bf16 v16bf;
typedef __attribute__((ext_vector_type(8)))  __bf16 v8bf;
typedef __attribute__((ext_vector_type(8)))  float  v8f;

// ---------------- helpers ----------------
__device__ __forceinline__ unsigned f2ord(float f) {
    unsigned b = __float_as_uint(f);
    return b ^ ((unsigned)((int)b >> 31) | 0x80000000u);
}
__device__ __forceinline__ float ord2f(unsigned u) {
    unsigned b = (u & 0x80000000u) ? (u ^ 0x80000000u) : ~u;
    return __uint_as_float(b);
}
__device__ __forceinline__ float wave_sum(float v) {
#pragma unroll
    for (int off = 16; off > 0; off >>= 1) v += __shfl_xor(v, off, 32);
    return v;
}
__device__ __forceinline__ float wave_max(float v) {
#pragma unroll
    for (int off = 16; off > 0; off >>= 1) v = fmaxf(v, __shfl_xor(v, off, 32));
    return v;
}
__device__ __forceinline__ void edge_sd(const int* __restrict__ ei, int e, int& s, int& d) {
    if (e < EE) { s = ei[e]; d = ei[EE + e]; }
    else        { s = e - EE; d = e - EE; }
}

// ---------------- prep: fp32 -> packed bf16 (8 elems/thread) ----------------
__global__ __launch_bounds__(256)
void cvt_bf16_kernel(const float* __restrict__ in, __bf16* __restrict__ out,
                     long long n) {
    const long long i = ((long long)blockIdx.x * blockDim.x + threadIdx.x) * 8;
    if (i >= n) return;
    v8bf o;
#pragma unroll
    for (int e = 0; e < 8; ++e) o[e] = (__bf16)in[i + e];
    *(v8bf*)(out + i) = o;
}

// ---------------- prep: transpose + convert W[K][N] -> WT[Npad][K] bf16 ----------------
__global__ __launch_bounds__(256)
void transpose_cvt_kernel(const float* __restrict__ in, __bf16* __restrict__ out,
                          int K, int N, int Npad) {
    const int tid = blockIdx.x * blockDim.x + threadIdx.x;
    if (tid >= Npad * K) return;
    const int n = tid / K;
    const int k = tid - n * K;
    out[tid] = (n < N) ? (__bf16)in[(size_t)k * N + n] : (__bf16)0.0f;
}

// ---------------- GEMM 1: XL = x@W1_l, XR = x@W1_r (both 50000x256) ----------------
// A shared between both products; all operands pre-packed bf16, K-contiguous.
__global__ __launch_bounds__(256)
void gemm1_kernel(const __bf16* __restrict__ xb, const __bf16* __restrict__ WlT,
                  const __bf16* __restrict__ WrT, float* __restrict__ XL,
                  float* __restrict__ XR) {
    const int lane  = threadIdx.x & 31;
    const int wid   = threadIdx.x >> 5;
    const int mtile = blockIdx.x;                 // 0..3124
    const int ntile = blockIdx.y * 8 + wid;       // 0..15
    const int h     = lane >> 4;                  // lane half
    const int lrow  = lane & 15;
    const int m     = mtile * 16 + lrow;
    const int n     = ntile * 16 + lrow;

    const __bf16* arow  = xb  + (size_t)m * F1;
    const __bf16* blrow = WlT + (size_t)n * F1;   // K-contiguous column of W
    const __bf16* brrow = WrT + (size_t)n * F1;

    v8f accL = {}; v8f accR = {};
#pragma unroll
    for (int k0 = 0; k0 < F1; k0 += 32) {
        // A 16x32 bf16: lanes0-15 K=0..15, lanes16-31 K=16..31
        v8bf a0 = *(const v8bf*)(arow + k0 + h * 16);
        v8bf a1 = *(const v8bf*)(arow + k0 + h * 16 + 8);
        v16bf a = __builtin_shufflevector(a0, a1, 0, 1, 2, 3, 4, 5, 6, 7,
                                          8, 9, 10, 11, 12, 13, 14, 15);
        // B 32x16 bf16: lanes0-15 K=0..15, lanes16-31 K=16..31 (col = lane&15)
        v16bf bl = *(const v16bf*)(blrow + k0 + h * 16);
        v16bf br = *(const v16bf*)(brrow + k0 + h * 16);
        accL = __builtin_amdgcn_wmma_f32_16x16x32_bf16(false, a, false, bl,
                                                       (short)0, accL, false, false);
        accR = __builtin_amdgcn_wmma_f32_16x16x32_bf16(false, a, false, br,
                                                       (short)0, accR, false, false);
    }
    // C/D: VGPR r -> M=r (lanes0-15) / M=r+8 (lanes16-31), N=lane&15
#pragma unroll
    for (int r = 0; r < 8; ++r) {
        const int mo = mtile * 16 + r + h * 8;
        XL[(size_t)mo * D1 + n] = accL[r];
        XR[(size_t)mo * D1 + n] = accR[r];
    }
}

// ---------------- GEMM 2: HL = h@W2_l, HR = h@W2_r (50000x40, tiles padded 48) ----------------
__global__ __launch_bounds__(64)
void gemm2_kernel(const __bf16* __restrict__ Hb, const __bf16* __restrict__ WlT,
                  const __bf16* __restrict__ WrT, float* __restrict__ HL,
                  float* __restrict__ HR) {
    const int lane  = threadIdx.x & 31;
    const int which = threadIdx.x >> 5;           // 0 -> L, 1 -> R
    const __bf16* WT = which ? WrT : WlT;
    float* Out       = which ? HR : HL;
    const int mtile = blockIdx.x;                 // 0..3124
    const int ntile = blockIdx.y;                 // 0..2
    const int h     = lane >> 4;
    const int lrow  = lane & 15;
    const int m     = mtile * 16 + lrow;
    const int n     = ntile * 16 + lrow;

    const __bf16* arow = Hb + (size_t)m * D1;
    const __bf16* brow = WT + (size_t)n * D1;     // padded rows 40..47 are zero

    v8f acc = {};
#pragma unroll
    for (int k0 = 0; k0 < D1; k0 += 32) {
        v8bf a0 = *(const v8bf*)(arow + k0 + h * 16);
        v8bf a1 = *(const v8bf*)(arow + k0 + h * 16 + 8);
        v16bf a = __builtin_shufflevector(a0, a1, 0, 1, 2, 3, 4, 5, 6, 7,
                                          8, 9, 10, 11, 12, 13, 14, 15);
        v16bf b = *(const v16bf*)(brow + k0 + h * 16);
        acc = __builtin_amdgcn_wmma_f32_16x16x32_bf16(false, a, false, b,
                                                      (short)0, acc, false, false);
    }
    if (n < D2) {
#pragma unroll
        for (int r = 0; r < 8; ++r) {
            const int mo = mtile * 16 + r + h * 8;
            Out[(size_t)mo * D2 + n] = acc[r];
        }
    }
}

// ---------------- layer-1 edge logits + running max (wave per edge) ----------------
__global__ __launch_bounds__(256)
void edge1_logits_kernel(const int* __restrict__ ei, const float* __restrict__ XL,
                         const float* __restrict__ XR, const float* __restrict__ att1,
                         float* __restrict__ EB, unsigned* __restrict__ NMX) {
    const int lane = threadIdx.x & 31;
    const int e = blockIdx.x * 8 + (threadIdx.x >> 5);
    if (e >= ETOT) return;
    int s, d; edge_sd(ei, e, s, d);
    const float* xl = XL + (size_t)s * D1;
    const float* xr = XR + (size_t)d * D1;
    __builtin_prefetch(xl + lane * 8, 0, 0);      // cover the full 1KB row
    __builtin_prefetch(xr + lane * 8, 0, 0);
#pragma unroll
    for (int hh = 0; hh < H1; ++hh) {
        const int c = hh * 32 + lane;
        float t = xl[c] + xr[c];
        t = t > 0.f ? t : NEG_SLOPE * t;
        const float v = wave_sum(t * att1[c]);
        if (lane == 0) {
            EB[(size_t)e * H1 + hh] = v;
            atomicMax(&NMX[(size_t)d * H1 + hh], f2ord(v));
        }
    }
}

// ---------------- layer-1 exp + denominator ----------------
__global__ __launch_bounds__(256)
void edge1_exp_kernel(const int* __restrict__ ei, float* __restrict__ EB,
                      const unsigned* __restrict__ NMX, float* __restrict__ DEN) {
    const long long i = (long long)blockIdx.x * blockDim.x + threadIdx.x;
    if (i >= (long long)ETOT * H1) return;
    const int e = (int)(i >> 3), hh = (int)(i & 7);
    int s, d; edge_sd(ei, e, s, d);
    const float mx = ord2f(NMX[(size_t)d * H1 + hh]);
    const float ex = __expf(EB[i] - mx);
    EB[i] = ex;
    atomicAdd(&DEN[(size_t)d * H1 + hh], ex);
}

// ---------------- layer-1 weighted scatter (wave per edge) ----------------
__global__ __launch_bounds__(256)
void edge1_scatter_kernel(const int* __restrict__ ei, const float* __restrict__ XL,
                          const float* __restrict__ EB, const float* __restrict__ DEN,
                          float* __restrict__ ACC) {
    const int lane = threadIdx.x & 31;
    const int e = blockIdx.x * 8 + (threadIdx.x >> 5);
    if (e >= ETOT) return;
    int s, d; edge_sd(ei, e, s, d);
    const float* xl = XL + (size_t)s * D1;
    float* outp = ACC + (size_t)d * D1;
    __builtin_prefetch(xl + lane * 8, 0, 0);
#pragma unroll
    for (int hh = 0; hh < H1; ++hh) {
        const float alpha = EB[(size_t)e * H1 + hh] / DEN[(size_t)d * H1 + hh];
        const int c = hh * 32 + lane;
        atomicAdd(&outp[c], alpha * xl[c]);
    }
}

// ---------------- bias + BN(eval) + ELU -> bf16 H ----------------
__global__ __launch_bounds__(256)
void post1_kernel(const float* __restrict__ ACC, __bf16* __restrict__ Hb,
                  const float* __restrict__ b1, const float* __restrict__ g,
                  const float* __restrict__ be, const float* __restrict__ mu,
                  const float* __restrict__ var) {
    const long long i = (long long)blockIdx.x * blockDim.x + threadIdx.x;
    if (i >= (long long)NM * D1) return;
    const int col = (int)(i & (D1 - 1));
    float v = ACC[i] + b1[col];
    v = (v - mu[col]) * (g[col] * rsqrtf(var[col] + 1e-5f)) + be[col];
    v = v > 0.f ? v : expm1f(v);
    Hb[i] = (__bf16)v;
}

// ---------------- layer-2 edge logits (1 head, 40 ch; wave per edge) ----------------
__global__ __launch_bounds__(256)
void edge2_logits_kernel(const int* __restrict__ ei, const float* __restrict__ HL,
                         const float* __restrict__ HR, const float* __restrict__ att2,
                         float* __restrict__ EB2, unsigned* __restrict__ NMX2) {
    const int lane = threadIdx.x & 31;
    const int e = blockIdx.x * 8 + (threadIdx.x >> 5);
    if (e >= ETOT) return;
    int s, d; edge_sd(ei, e, s, d);
    const float* hl = HL + (size_t)s * D2;
    const float* hr = HR + (size_t)d * D2;
    float t = hl[lane] + hr[lane];
    t = t > 0.f ? t : NEG_SLOPE * t;
    float v = t * att2[lane];
    if (lane < 8) {
        const int c = 32 + lane;
        float t2 = hl[c] + hr[c];
        t2 = t2 > 0.f ? t2 : NEG_SLOPE * t2;
        v += t2 * att2[c];
    }
    v = wave_sum(v);
    if (lane == 0) {
        EB2[e] = v;
        atomicMax(&NMX2[d], f2ord(v));
    }
}

__global__ __launch_bounds__(256)
void edge2_exp_kernel(const int* __restrict__ ei, float* __restrict__ EB2,
                      const unsigned* __restrict__ NMX2, float* __restrict__ DEN2) {
    const int e = blockIdx.x * blockDim.x + threadIdx.x;
    if (e >= ETOT) return;
    int s, d; edge_sd(ei, e, s, d);
    const float ex = __expf(EB2[e] - ord2f(NMX2[d]));
    EB2[e] = ex;
    atomicAdd(&DEN2[d], ex);
}

__global__ __launch_bounds__(256)
void edge2_scatter_kernel(const int* __restrict__ ei, const float* __restrict__ HL,
                          const float* __restrict__ EB2, const float* __restrict__ DEN2,
                          float* __restrict__ ACC2) {
    const int lane = threadIdx.x & 31;
    const int e = blockIdx.x * 8 + (threadIdx.x >> 5);
    if (e >= ETOT) return;
    int s, d; edge_sd(ei, e, s, d);
    const float alpha = EB2[e] / DEN2[d];
    const float* hl = HL + (size_t)s * D2;
    float* outp = ACC2 + (size_t)d * D2;
    atomicAdd(&outp[lane], alpha * hl[lane]);
    if (lane < 8) atomicAdd(&outp[32 + lane], alpha * hl[32 + lane]);
}

// ---------------- bias + log_softmax (wave per row) ----------------
__global__ __launch_bounds__(256)
void final_kernel(const float* __restrict__ ACC2, const float* __restrict__ b2,
                  float* __restrict__ out) {
    const int lane = threadIdx.x & 31;
    const int r = blockIdx.x * 8 + (threadIdx.x >> 5);
    if (r >= NM) return;
    const float* row = ACC2 + (size_t)r * D2;
    const float v0 = row[lane] + b2[lane];
    const float v1 = (lane < 8) ? row[32 + lane] + b2[32 + lane] : -3.402823e38f;
    const float mx = wave_max(fmaxf(v0, v1));
    float se = __expf(v0 - mx) + ((lane < 8) ? __expf(v1 - mx) : 0.f);
    se = wave_sum(se);
    const float ls = mx + __logf(se);
    out[(size_t)r * D2 + lane] = v0 - ls;
    if (lane < 8) out[(size_t)r * D2 + 32 + lane] = v1 - ls;
}

// ---------------- host launch ----------------
extern "C" void kernel_launch(void* const* d_in, const int* in_sizes, int n_in,
                              void* d_out, int out_size, void* d_ws, size_t ws_size,
                              hipStream_t stream) {
    const float* x    = (const float*)d_in[0];
    const int*   ei   = (const int*)d_in[1];
    const float* W1l  = (const float*)d_in[2];
    const float* W1r  = (const float*)d_in[3];
    const float* att1 = (const float*)d_in[4];
    const float* b1   = (const float*)d_in[5];
    const float* bng  = (const float*)d_in[6];
    const float* bnb  = (const float*)d_in[7];
    const float* bnm  = (const float*)d_in[8];
    const float* bnv  = (const float*)d_in[9];
    const float* W2l  = (const float*)d_in[10];
    const float* W2r  = (const float*)d_in[11];
    const float* att2 = (const float*)d_in[12];
    const float* b2   = (const float*)d_in[13];
    float* out = (float*)d_out;

    // ---------------- workspace carve-up ----------------
    char* p = (char*)d_ws;
    float* XL  = (float*)p; p += (size_t)NM * D1 * 4;
    float* XR  = (float*)p; p += (size_t)NM * D1 * 4;
    float* HL  = (float*)p; p += (size_t)NM * D2 * 4;
    float* HR  = (float*)p; p += (size_t)NM * D2 * 4;
    float* EB  = (float*)p; p += (size_t)ETOT * H1 * 4;
    float* EB2 = (float*)p; p += (size_t)ETOT * 4;
    // ---- contiguous zero-init region ----
    char* zbase = p;
    float* ACC  = (float*)p; p += (size_t)NM * D1 * 4;
    float* ACC2 = (float*)p; p += (size_t)NM * D2 * 4;
    float* DEN  = (float*)p; p += (size_t)NM * H1 * 4;
    float* DEN2 = (float*)p; p += (size_t)NM * 4;
    unsigned* NMX  = (unsigned*)p; p += (size_t)NM * H1 * 4;  // ordered-uint, 0 == -inf
    unsigned* NMX2 = (unsigned*)p; p += (size_t)NM * 4;
    const size_t zero_bytes = (size_t)(p - zbase);
    // ---- bf16 staging buffers ----
    __bf16* xb   = (__bf16*)p; p += (size_t)NM * F1 * 2;
    __bf16* W1lT = (__bf16*)p; p += (size_t)D1 * F1 * 2;
    __bf16* W1rT = (__bf16*)p; p += (size_t)D1 * F1 * 2;
    __bf16* Hb   = (__bf16*)p; p += (size_t)NM * D1 * 2;
    __bf16* W2lT = (__bf16*)p; p += (size_t)D2P * D1 * 2;
    __bf16* W2rT = (__bf16*)p; p += (size_t)D2P * D1 * 2;

    hipMemsetAsync(zbase, 0, zero_bytes, stream);

    // ---------------- prep: bf16 conversion / weight transposes ----------------
    const long long nx = (long long)NM * F1;
    cvt_bf16_kernel<<<(int)((nx / 8 + 255) / 256), 256, 0, stream>>>(x, xb, nx);
    transpose_cvt_kernel<<<(D1 * F1 + 255) / 256, 256, 0, stream>>>(W1l, W1lT, F1, D1, D1);
    transpose_cvt_kernel<<<(D1 * F1 + 255) / 256, 256, 0, stream>>>(W1r, W1rT, F1, D1, D1);
    transpose_cvt_kernel<<<(D2P * D1 + 255) / 256, 256, 0, stream>>>(W2l, W2lT, D1, D2, D2P);
    transpose_cvt_kernel<<<(D2P * D1 + 255) / 256, 256, 0, stream>>>(W2r, W2rT, D1, D2, D2P);

    // ---------------- layer 1 ----------------
    gemm1_kernel<<<dim3(NM / 16, 2), 256, 0, stream>>>(xb, W1lT, W1rT, XL, XR);
    edge1_logits_kernel<<<(ETOT + 7) / 8, 256, 0, stream>>>(ei, XL, XR, att1, EB, NMX);
    edge1_exp_kernel<<<(int)(((long long)ETOT * H1 + 255) / 256), 256, 0, stream>>>(ei, EB, NMX, DEN);
    edge1_scatter_kernel<<<(ETOT + 7) / 8, 256, 0, stream>>>(ei, XL, EB, DEN, ACC);
    post1_kernel<<<(int)(((long long)NM * D1 + 255) / 256), 256, 0, stream>>>(ACC, Hb, b1, bng, bnb, bnm, bnv);

    // ---------------- layer 2 ----------------
    gemm2_kernel<<<dim3(NM / 16, 3), 64, 0, stream>>>(Hb, W2lT, W2rT, HL, HR);
    edge2_logits_kernel<<<(ETOT + 7) / 8, 256, 0, stream>>>(ei, HL, HR, att2, EB2, NMX2);
    edge2_exp_kernel<<<(ETOT + 255) / 256, 256, 0, stream>>>(ei, EB2, NMX2, DEN2);
    edge2_scatter_kernel<<<(ETOT + 7) / 8, 256, 0, stream>>>(ei, HL, EB2, DEN2, ACC2);
    final_kernel<<<(NM + 7) / 8, 256, 0, stream>>>(ACC2, b2, out);
}
